// GAT_NET_30185030156397
// MI455X (gfx1250) — matrix operand order repaired and checked by reading, not compiled, not run on hardware
//
#include <hip/hip_runtime.h>
#include <hip/hip_bf16.h>
#include <math.h>

typedef _Float16 v16h __attribute__((ext_vector_type(16)));
typedef _Float16 v8h  __attribute__((ext_vector_type(8)));
typedef float    v8f  __attribute__((ext_vector_type(8)));

#define TPB 256

// ---------------------------------------------------------------- utilities

__global__ void fill_u32_kernel(unsigned* __restrict__ p, unsigned v, long n) {
    long t = (long)blockIdx.x * blockDim.x + threadIdx.x;
    if (t < n) p[t] = v;
}

__global__ void cvt_f16_kernel(const float* __restrict__ x, _Float16* __restrict__ y, long n) {
    long t = (long)blockIdx.x * blockDim.x + threadIdx.x;
    if (t < n) y[t] = (_Float16)x[t];
}

// W is [K x Ncols] row-major fp32; produce Wt [Npad x K] f16 (transposed, zero-padded cols)
__global__ void wconv_t_kernel(const float* __restrict__ W, _Float16* __restrict__ Wt,
                               int K, int Ncols, int Npad) {
    int t = blockIdx.x * blockDim.x + threadIdx.x;
    if (t >= Npad * K) return;
    int n = t / K, k = t % K;
    Wt[t] = (n < Ncols) ? (_Float16)W[(size_t)k * Ncols + n] : (_Float16)0.0f;
}

// ---------------------------------------------------------------- WMMA GEMM
// C[M x ldc] = A[M x K](f16) @ Bt[ldc x K]^T (f16, stored N-major = B transposed)
// One wave computes a 16x16 tile; 4 waves per block along M.
__global__ void gemm_wmma_kernel(const _Float16* __restrict__ A,
                                 const _Float16* __restrict__ Bt,
                                 float* __restrict__ C,
                                 int M, int K, int ldc) {
    const int lane  = threadIdx.x & 31;
    const int wave  = threadIdx.x >> 5;
    const int mtile = blockIdx.x * (blockDim.x >> 5) + wave;
    if (mtile * 16 >= M) return;                 // wave-uniform exit (EXEC stays all-ones)
    const int m0 = mtile * 16;
    const int n0 = blockIdx.y * 16;
    const int hi = lane >> 4;                    // 0: lanes 0-15, 1: lanes 16-31
    const int l  = lane & 15;
    const size_t arow = (size_t)(m0 + l) * K;
    const size_t brow = (size_t)(n0 + l) * K;
    const int kA = hi ? 8 : 0;                   // A: lanes<16 hold K {0-7,16-23}
    const int kB = hi ? 8 : 0;                   // B: lanes<16 hold K {0-7,16-23}

    v8f acc{};
    for (int k0 = 0; k0 < K; k0 += 32) {
        const v8h alo = *(const v8h*)(A  + arow + k0 + kA);
        const v8h ahi = *(const v8h*)(A  + arow + k0 + kA + 16);
        const v8h blo = *(const v8h*)(Bt + brow + k0 + kB);
        const v8h bhi = *(const v8h*)(Bt + brow + k0 + kB + 16);
        v16h av, bv;
#pragma unroll
        for (int j = 0; j < 8; ++j) {
            av[j] = alo[j]; av[j + 8] = ahi[j];
            bv[j] = blo[j]; bv[j + 8] = bhi[j];
        }
        acc = __builtin_amdgcn_wmma_f32_16x16x32_f16(
                  false, av, false, bv, (short)0, acc, false, false);
    }
#pragma unroll
    for (int r = 0; r < 8; ++r)
        C[(size_t)(m0 + r + hi * 8) * ldc + n0 + l] = acc[r];
}

// ---------------------------------------------------------------- attention

__global__ void attn_kernel(const float* __restrict__ Hb,
                            const float* __restrict__ aS, const float* __restrict__ aD,
                            float* __restrict__ es, float* __restrict__ ed,
                            long Nn, int H, int C, int ldh) {
    long t = (long)blockIdx.x * blockDim.x + threadIdx.x;
    if (t >= Nn * H) return;
    long n = t / H; int h = (int)(t % H);
    const float* row = Hb + (size_t)n * ldh + h * C;
    float s = 0.f, d = 0.f;
    for (int c = 0; c < C; ++c) { float v = row[c]; s += v * aS[h * C + c]; d += v * aD[h * C + c]; }
    es[t] = s; ed[t] = d;
}

__device__ __forceinline__ void edge_sd(const int* src, const int* dst, long eid, long E,
                                        int& s, int& d) {
    if (eid < E) { s = src[eid]; d = dst[eid]; }
    else         { s = d = (int)(eid - E); }     // self loops appended
}
__device__ __forceinline__ float lrelu(float x) { return x > 0.f ? x : 0.2f * x; }
__device__ __forceinline__ unsigned ford(float x) {
    unsigned u = __float_as_uint(x);
    return (u & 0x80000000u) ? ~u : (u | 0x80000000u);
}
__device__ __forceinline__ float forddec(unsigned o) {
    unsigned u = (o & 0x80000000u) ? (o & 0x7FFFFFFFu) : ~o;
    return __uint_as_float(u);
}

__global__ void edge_max_kernel(const int* __restrict__ src, const int* __restrict__ dst,
                                long E, long Etot, int H,
                                const float* __restrict__ es, const float* __restrict__ ed,
                                unsigned* __restrict__ mx) {
    long t = (long)blockIdx.x * blockDim.x + threadIdx.x;
    if (t >= Etot * H) return;
    long eid = t / H; int h = (int)(t % H);
    int s, d; edge_sd(src, dst, eid, E, s, d);
    float e = lrelu(es[(size_t)s * H + h] + ed[(size_t)d * H + h]);
    atomicMax(&mx[(size_t)d * H + h], ford(e));
}

__global__ void edge_den_kernel(const int* __restrict__ src, const int* __restrict__ dst,
                                long E, long Etot, int H,
                                const float* __restrict__ es, const float* __restrict__ ed,
                                const unsigned* __restrict__ mx, float* __restrict__ den) {
    long t = (long)blockIdx.x * blockDim.x + threadIdx.x;
    if (t >= Etot * H) return;
    long eid = t / H; int h = (int)(t % H);
    int s, d; edge_sd(src, dst, eid, E, s, d);
    size_t di = (size_t)d * H + h;
    float e  = lrelu(es[(size_t)s * H + h] + ed[di]);
    atomicAdd(&den[di], __expf(e - forddec(mx[di])));
}

__global__ void edge_alpha_kernel(const int* __restrict__ src, const int* __restrict__ dst,
                                  long E, long Etot, int H,
                                  const float* __restrict__ es, const float* __restrict__ ed,
                                  const unsigned* __restrict__ mx, const float* __restrict__ den,
                                  float* __restrict__ alpha) {
    long t = (long)blockIdx.x * blockDim.x + threadIdx.x;
    if (t >= Etot * H) return;
    long eid = t / H; int h = (int)(t % H);
    int s, d; edge_sd(src, dst, eid, E, s, d);
    size_t di = (size_t)d * H + h;
    float e = lrelu(es[(size_t)s * H + h] + ed[di]);
    alpha[t] = __expf(e - forddec(mx[di])) / den[di];
}

__global__ void edge_scatter_kernel(const int* __restrict__ src, const int* __restrict__ dst,
                                    long E, long Etot, int H, int C, int ldh,
                                    const float* __restrict__ alpha,
                                    const float* __restrict__ Hbuf, float* __restrict__ out) {
    long HC = (long)H * C;
    long t = (long)blockIdx.x * blockDim.x + threadIdx.x;
    if (t >= Etot * HC) return;
    long eid = t / HC; int f = (int)(t % HC); int h = f / C;
    int s, d; edge_sd(src, dst, eid, E, s, d);
    atomicAdd(&out[(size_t)d * HC + f], alpha[eid * H + h] * Hbuf[(size_t)s * ldh + f]);
}

// ---------------------------------------------------------------- batchnorm

__global__ void bn_stats_kernel(const float* __restrict__ X, int Nrows, int F,
                                float* __restrict__ mu, float* __restrict__ rs) {
    __shared__ float s1[TPB], s2[TPB];
    int f = blockIdx.x;
    float a = 0.f, b = 0.f;
    for (int r = threadIdx.x; r < Nrows; r += blockDim.x) {
        float v = X[(size_t)r * F + f];
        a += v; b += v * v;
    }
    s1[threadIdx.x] = a; s2[threadIdx.x] = b;
    __syncthreads();
    for (int s = TPB / 2; s > 0; s >>= 1) {
        if (threadIdx.x < s) {
            s1[threadIdx.x] += s1[threadIdx.x + s];
            s2[threadIdx.x] += s2[threadIdx.x + s];
        }
        __syncthreads();
    }
    if (threadIdx.x == 0) {
        float m   = s1[0] / Nrows;
        float var = s2[0] / Nrows - m * m;
        mu[f] = m;
        rs[f] = rsqrtf(var + 1e-5f);
    }
}

__global__ void bn_apply_kernel(float* __restrict__ X,
                                const float* __restrict__ mu, const float* __restrict__ rs,
                                const float* __restrict__ g,  const float* __restrict__ be,
                                long n, int F, int relu, _Float16* __restrict__ Xh) {
    long t = (long)blockIdx.x * blockDim.x + threadIdx.x;
    if (t >= n) return;
    int f = (int)(t % F);
    float v = (X[t] - mu[f]) * rs[f] * g[f] + be[f];
    if (relu) v = fmaxf(v, 0.f);
    X[t] = v;
    if (Xh) Xh[t] = (_Float16)v;
}

// ---------------------------------------------------------------- pool + head

__global__ void pool_kernel(const float* __restrict__ Hb, const int* __restrict__ batch,
                            float* __restrict__ pool, float* __restrict__ cnt, long Nn) {
    long t = (long)blockIdx.x * blockDim.x + threadIdx.x;
    if (t >= Nn * 20) return;
    long n = t / 20; int c = (int)(t % 20);
    int g = batch[n];
    atomicAdd(&pool[g * 20 + c], Hb[t]);
    if (c == 0) atomicAdd(&cnt[g], 1.0f);
}

__global__ void classify_kernel(const float* __restrict__ pool, const float* __restrict__ cnt,
                                const float* __restrict__ linW, const float* __restrict__ linb,
                                float* __restrict__ out, int G) {
    int g = blockIdx.x * blockDim.x + threadIdx.x;
    if (g >= G) return;
    float inv = 1.0f / fmaxf(cnt[g], 1.0f);
    float lg[6]; float mx = -1e30f;
    for (int c = 0; c < 6; ++c) {
        float s = linb[c];
        for (int j = 0; j < 20; ++j) s += pool[g * 20 + j] * inv * linW[j * 6 + c];
        lg[c] = s; mx = fmaxf(mx, s);
    }
    float den = 0.f;
    for (int c = 0; c < 6; ++c) { lg[c] = __expf(lg[c] - mx); den += lg[c]; }
    for (int c = 0; c < 6; ++c) out[g * 6 + c] = lg[c] / den;
}

// ---------------------------------------------------------------- host side

static inline unsigned nblk(long n) { return (unsigned)((n + TPB - 1) / TPB); }

static void launch_gemm(const _Float16* A, const _Float16* Bt, float* C,
                        long M, int K, int Ncols, hipStream_t st) {
    long mt = (M + 15) / 16;
    dim3 grid((unsigned)((mt + 3) / 4), (unsigned)(Ncols / 16));
    gemm_wmma_kernel<<<grid, 128, 0, st>>>(A, Bt, C, (int)M, K, Ncols);
}

static void gat_aggregate(const float* Hgemm, int H, int C, int ldh,
                          const float* aS, const float* aD,
                          const int* src, const int* dst, long E, long N,
                          float* ES, float* ED, unsigned* MX, float* DEN, float* ALPHA,
                          float* OUT, hipStream_t st) {
    const long Etot = E + N;
    const long HC = (long)H * C;
    attn_kernel<<<nblk(N * H), TPB, 0, st>>>(Hgemm, aS, aD, ES, ED, N, H, C, ldh);
    fill_u32_kernel<<<nblk(N * H),  TPB, 0, st>>>(MX, 0u, N * H);
    fill_u32_kernel<<<nblk(N * H),  TPB, 0, st>>>((unsigned*)DEN, 0u, N * H);
    fill_u32_kernel<<<nblk(N * HC), TPB, 0, st>>>((unsigned*)OUT, 0u, N * HC);
    edge_max_kernel  <<<nblk(Etot * H),  TPB, 0, st>>>(src, dst, E, Etot, H, ES, ED, MX);
    edge_den_kernel  <<<nblk(Etot * H),  TPB, 0, st>>>(src, dst, E, Etot, H, ES, ED, MX, DEN);
    edge_alpha_kernel<<<nblk(Etot * H),  TPB, 0, st>>>(src, dst, E, Etot, H, ES, ED, MX, DEN, ALPHA);
    edge_scatter_kernel<<<nblk(Etot * HC), TPB, 0, st>>>(src, dst, E, Etot, H, C, ldh,
                                                         ALPHA, Hgemm, OUT);
}

extern "C" void kernel_launch(void* const* d_in, const int* in_sizes, int n_in,
                              void* d_out, int out_size, void* d_ws, size_t ws_size,
                              hipStream_t stream) {
    const float* x     = (const float*)d_in[0];
    const int*   ei    = (const int*)d_in[1];
    const int*   batch = (const int*)d_in[2];
    const float* W1  = (const float*)d_in[3];
    const float* as1 = (const float*)d_in[4];
    const float* ad1 = (const float*)d_in[5];
    const float* g1  = (const float*)d_in[7];
    const float* be1 = (const float*)d_in[8];
    const float* W2  = (const float*)d_in[9];
    const float* as2 = (const float*)d_in[10];
    const float* ad2 = (const float*)d_in[11];
    const float* g2  = (const float*)d_in[13];
    const float* be2 = (const float*)d_in[14];
    const float* W3  = (const float*)d_in[15];
    const float* as3 = (const float*)d_in[16];
    const float* ad3 = (const float*)d_in[17];
    const float* g3  = (const float*)d_in[19];
    const float* be3 = (const float*)d_in[20];
    const float* linW = (const float*)d_in[21];
    const float* linb = (const float*)d_in[22];

    const long N = in_sizes[0] / 128;
    const long E = in_sizes[1] / 2;
    const int* src = ei;
    const int* dst = ei + E;
    const int  G = 64;

    // ---- workspace carve-out
    size_t off = 0;
    auto alloc = [&](size_t bytes) -> void* {
        size_t o = (off + 255) & ~(size_t)255;
        off = o + bytes;
        return (void*)((char*)d_ws + o);
    };
    float*     B0    = (float*)    alloc((size_t)N * 256 * 4);  // gemm output h
    float*     B1    = (float*)    alloc((size_t)N * 256 * 4);  // aggregated / BN output
    _Float16*  B1h   = (_Float16*) alloc((size_t)N * 256 * 2);  // f16 gemm input
    _Float16*  W1t   = (_Float16*) alloc((size_t)256 * 128 * 2);
    _Float16*  W2t   = (_Float16*) alloc((size_t)128 * 256 * 2);
    _Float16*  W3t   = (_Float16*) alloc((size_t)32  * 128 * 2);
    float*     ES    = (float*)    alloc((size_t)N * 4 * 4);
    float*     ED    = (float*)    alloc((size_t)N * 4 * 4);
    unsigned*  MX    = (unsigned*) alloc((size_t)N * 4 * 4);
    float*     DEN   = (float*)    alloc((size_t)N * 4 * 4);
    float*     ALPHA = (float*)    alloc((size_t)(E + N) * 4 * 4);
    float*     MU    = (float*)    alloc(256 * 4);
    float*     RSIG  = (float*)    alloc(256 * 4);
    float*     POOL  = (float*)    alloc((size_t)G * 20 * 4);
    float*     CNT   = (float*)    alloc((size_t)G * 4);
    (void)ws_size; (void)n_in; (void)out_size;

    // ---- prep: f16 activations + transposed f16 weights
    cvt_f16_kernel<<<nblk(N * 128), TPB, 0, stream>>>(x, B1h, N * 128);
    wconv_t_kernel<<<nblk(256 * 128), TPB, 0, stream>>>(W1, W1t, 128, 256, 256);
    wconv_t_kernel<<<nblk(128 * 256), TPB, 0, stream>>>(W2, W2t, 256, 128, 128);
    wconv_t_kernel<<<nblk(32  * 128), TPB, 0, stream>>>(W3, W3t, 128, 20, 32);

    // ---- layer 1: 128 -> 4x64 (concat 256)   (GAT bias cancels inside BN)
    launch_gemm(B1h, W1t, B0, N, 128, 256, stream);
    gat_aggregate(B0, 4, 64, 256, as1, ad1, src, dst, E, N, ES, ED, MX, DEN, ALPHA, B1, stream);
    bn_stats_kernel<<<256, TPB, 0, stream>>>(B1, (int)N, 256, MU, RSIG);
    bn_apply_kernel<<<nblk(N * 256), TPB, 0, stream>>>(B1, MU, RSIG, g1, be1,
                                                       N * 256, 256, 1, B1h);

    // ---- layer 2: 256 -> 4x32 (concat 128)
    launch_gemm(B1h, W2t, B0, N, 256, 128, stream);
    gat_aggregate(B0, 4, 32, 128, as2, ad2, src, dst, E, N, ES, ED, MX, DEN, ALPHA, B1, stream);
    bn_stats_kernel<<<128, TPB, 0, stream>>>(B1, (int)N, 128, MU, RSIG);
    bn_apply_kernel<<<nblk(N * 128), TPB, 0, stream>>>(B1, MU, RSIG, g2, be2,
                                                       N * 128, 128, 1, B1h);

    // ---- layer 3: 128 -> 1x20 (N padded to 32 in gemm; cols 20..31 are zero weights)
    launch_gemm(B1h, W3t, B0, N, 128, 32, stream);
    gat_aggregate(B0, 1, 20, 32, as3, ad3, src, dst, E, N, ES, ED, MX, DEN, ALPHA, B1, stream);
    bn_stats_kernel<<<20, TPB, 0, stream>>>(B1, (int)N, 20, MU, RSIG);
    bn_apply_kernel<<<nblk(N * 20), TPB, 0, stream>>>(B1, MU, RSIG, g3, be3,
                                                      N * 20, 20, 0, (_Float16*)nullptr);

    // ---- global mean pool + classifier + softmax
    fill_u32_kernel<<<nblk(G * 20), TPB, 0, stream>>>((unsigned*)POOL, 0u, (long)G * 20);
    fill_u32_kernel<<<nblk(G),      TPB, 0, stream>>>((unsigned*)CNT,  0u, (long)G);
    pool_kernel<<<nblk(N * 20), TPB, 0, stream>>>(B1, batch, POOL, CNT, N);
    classify_kernel<<<nblk(G), TPB, 0, stream>>>(POOL, CNT, linW, linb, (float*)d_out, G);
}
